// NewCustomConv2_8770323218907
// MI455X (gfx1250) — 8-bit matrix operand order repaired and checked by reading, compiled, not run on hardware
//
#include <hip/hip_runtime.h>

typedef __attribute__((ext_vector_type(16))) int   v16i;
typedef __attribute__((ext_vector_type(8)))  float v8f;

__device__ __forceinline__ int digit_serial(int x, int k, int nb) {
    int s = (k > 0) - (k < 0);
    int w = k < 0 ? -k : k;
    int d = 0;
    for (int t = 0; t < nb; ++t) {
        d += x * (w & 3) * s;
        w >>= 2;
        d = d / 4;
    }
    return d;
}

__device__ __forceinline__ unsigned enc_e4m3(int v) {
    if (v == 0) return 0u;
    unsigned s = 0u, m;
    if (v < 0) { s = 0x80u; m = (unsigned)(-v); } else { m = (unsigned)v; }
    int e = 31 - __clz((int)m);
    unsigned frac = (m << (3 - e)) & 7u;
    return s | ((unsigned)(e + 7) << 3) | frac;
}

__global__ void quant_inputs(const float* __restrict__ in,
                             unsigned char* __restrict__ x8, int n) {
    int i = blockIdx.x * blockDim.x + threadIdx.x;
    if (i < n) x8[i] = (unsigned char)(int)in[i];
}

__global__ void build_wf(const float* __restrict__ kern,
                         const int* __restrict__ bitsp,
                         unsigned int* __restrict__ wf) {
    int idx   = blockIdx.x * blockDim.x + threadIdx.x;
    int v     = idx & 15;
    int lane  = (idx >> 4) & 31;
    int ft    = (idx >> 9) & 3;
    int chunk = idx >> 11;
    int tap   = chunk >> 3;
    int cg    = chunk & 7;
    int f     = ft * 16 + (lane & 15);
    int nb    = bitsp[0] >> 1;
    unsigned int word = 0;
    for (int b = 0; b < 4; ++b) {
        int K  = ((v & 8) << 3) + ((v >> 1) & 3) * 16 + (v & 1) * 4 + ((lane >= 16) ? 8 : 0) + b;
        int c  = cg * 8 + (K >> 4);
        int dg = K & 15;
        int kv = (int)kern[(tap * 64 + c) * 64 + f];
        word |= enc_e4m3(digit_serial(dg, kv, nb)) << (8 * b);
    }
    wf[idx] = word;
}

__global__ void expand_onehot(const unsigned char* __restrict__ x8,
                              unsigned int* __restrict__ xexp) {
    int idx   = blockIdx.x * blockDim.x + threadIdx.x;
    int pixel = idx >> 8;
    int wi    = idx & 255;
    int cg    = wi >> 5;
    int half  = (wi >> 4) & 1;
    int v     = wi & 15;
    unsigned int word = 0;
    if (pixel < 8192) {
        for (int b = 0; b < 4; ++b) {
            int K  = ((v & 8) << 3) + ((v >> 1) & 3) * 16 + (v & 1) * 4
                     + half * 8 + b;
            int xv = x8[pixel * 64 + cg * 8 + (K >> 4)];
            word |= ((xv == (K & 15)) ? 0x38u : 0u) << (8 * b);
        }
    }
    xexp[idx] = word;
}

__device__ __forceinline__ void stage_issue(const unsigned char* g,
                                            unsigned lds, int tid) {
    unsigned off = lds + (unsigned)tid * 128u;
    unsigned long long ga =
        (unsigned long long)(uintptr_t)g + (unsigned long long)tid * 128u;
#pragma unroll
    for (int i = 0; i < 8; ++i) {
        unsigned o = off + (unsigned)(i * 16);
        unsigned long long a = ga + (unsigned long long)(i * 16);
        asm volatile("global_load_async_to_lds_b128 %0, %1, off"
                     :: "v"(o), "v"(a) : "memory");
    }
}

__global__ __launch_bounds__(256) void conv_bitserial_wmma(
    const unsigned char* __restrict__ xexp, const unsigned char* __restrict__ wf,
    const float* __restrict__ bias, float* __restrict__ out) {
    __shared__ unsigned char lbuf[2][32768];

    const int tid   = threadIdx.x;
    const int lane  = tid & 31;
    const int wave  = tid >> 5;
    const int Mtile = blockIdx.x * 128 + wave * 16;
    const int p     = Mtile + (lane & 15);
    const int wc    = p & 31;
    const int hr    = (p >> 5) & 31;
    const int bi    = p >> 10;
    const int hi8   = (lane < 16) ? 0 : 8;
    const int hi64  = (lane < 16) ? 0 : 64;
    const unsigned lds0 = (unsigned)(uintptr_t)&lbuf[0][0];
    const unsigned lds1 = (unsigned)(uintptr_t)&lbuf[1][0];

    v8f acc[4];
#pragma unroll
    for (int i = 0; i < 4; ++i)
        acc[i] = (v8f){0.f, 0.f, 0.f, 0.f, 0.f, 0.f, 0.f, 0.f};

    stage_issue(wf, lds0, tid);

    for (int ti = 0; ti < 3; ++ti) {
        for (int tj = 0; tj < 3; ++tj) {
            const int tap = ti * 3 + tj;
            const int hs = hr + ti - 1, ws = wc + tj - 1;
            const bool inb = (hs >= 0) && (hs < 32) && (ws >= 0) && (ws < 32);
            const int ppix = inb ? ((bi * 32 + hs) * 32 + ws) : 8192;
            const unsigned char* aptr = xexp + (unsigned)ppix * 1024u + hi64;

#pragma unroll
            for (int half = 0; half < 2; ++half) {
                const int s = tap * 2 + half;
                if (s < 17) {
                    stage_issue(wf + (unsigned)(s + 1) * 32768u,
                                (s & 1) ? lds0 : lds1, tid);
                    asm volatile("s_wait_asynccnt 0x8" ::: "memory");
                } else {
                    asm volatile("s_wait_asynccnt 0x0" ::: "memory");
                }
                __syncthreads();
                const unsigned char* bufb = (s & 1) ? &lbuf[1][0] : &lbuf[0][0];

#pragma unroll
                for (int lc = 0; lc < 4; ++lc) {
                    const int cg = half * 4 + lc;
                    v16i A = *(const v16i*)(aptr + cg * 128);
                    const unsigned char* bb = bufb + lc * 8192 + lane * 64;
#pragma unroll
                    for (int ft = 0; ft < 4; ++ft) {
                        v16i B = *(const v16i*)(bb + ft * 2048);
                        acc[ft] = __builtin_amdgcn_wmma_f32_16x16x128_fp8_fp8(
                            A, B, (short)0, acc[ft],
                             false,  false);
                    }
                }
                __syncthreads();
            }
        }
    }

    const int col = lane & 15;
#pragma unroll
    for (int ft = 0; ft < 4; ++ft) {
        const float bv = bias[ft * 16 + col];
#pragma unroll
        for (int r = 0; r < 8; ++r) {
            float val = acc[ft][r] + bv;
            out[(Mtile + hi8 + r) * 64 + ft * 16 + col] = val > 0.f ? val : 0.f;
        }
    }
}

extern "C" void kernel_launch(void* const* d_in, const int* in_sizes, int n_in,
                              void* d_out, int out_size, void* d_ws, size_t ws_size,
                              hipStream_t stream) {
    const float* inp   = (const float*)d_in[0];
    const float* kern  = (const float*)d_in[1];
    const float* bias  = (const float*)d_in[2];
    const int*   bitsp = (const int*)d_in[3];
    float* out = (float*)d_out;

    unsigned char* x8   = (unsigned char*)d_ws;
    unsigned char* wf   = (unsigned char*)d_ws + 524288;
    unsigned char* xexp = (unsigned char*)d_ws + 1114112;

    const int n = in_sizes[0];
    quant_inputs<<<(n + 255) / 256, 256, 0, stream>>>(inp, x8, n);
    build_wf<<<576, 256, 0, stream>>>(kern, bitsp, (unsigned int*)wf);
    expand_onehot<<<8193, 256, 0, stream>>>(x8, (unsigned int*)xexp);
    conv_bitserial_wmma<<<64, 256, 0, stream>>>(xexp, wf, bias, out);
}
